// MixtralDecoderLayer_49667001811797
// MI455X (gfx1250) — matrix operand order repaired and checked by reading, not compiled, not run on hardware
//
#include <hip/hip_runtime.h>
#include <hip/hip_bf16.h>

// ---------------------------------------------------------------------------
// MI455X / gfx1250 Mixtral decoder layer.
// bf16 WMMA (v_wmma_f32_16x16x32_bf16) for every matmul, fp32 accumulate.
// fp32 weights stream from HBM once (bandwidth-bound: ~1.4GB MoE weights at
// 23.3 TB/s sets the floor), converted to bf16 in LDS via batched float4
// loads (all in flight together) + packed bf16x4 LDS stores. LDS rows padded
// to non-power-of-two strides to spread fragment reads across the 64 banks.
// CDNA5 paths: v_wmma, global_prefetch_b8, global_load_async_to_lds_b128.
// ---------------------------------------------------------------------------

typedef __attribute__((ext_vector_type(16))) __bf16 v16bf;
typedef __attribute__((ext_vector_type(4)))  __bf16 v4bf;
typedef __attribute__((ext_vector_type(8)))  float  v8f;
typedef __attribute__((ext_vector_type(4)))  float  v4f;

#define LDA 40       // bf16 tile row stride (80B = 20 dwords: 16 distinct banks, 16B aligned)
#define LDQ 136      // attention bf16 tile row stride (272B = 68 dwords -> stride 4 banks)
#define LDS_SC 1028  // score row stride in floats (4112B -> stride 4 banks)

__device__ __forceinline__ v8f wmma_bf16(v16bf a, v16bf b, v8f c) {
  return __builtin_amdgcn_wmma_f32_16x16x32_bf16(
      /*neg_a=*/false, a, /*neg_b=*/false, b,
      /*c_mod=*/(short)0, c, /*reuse_a=*/false, /*reuse_b=*/false);
}

__device__ __forceinline__ v4bf cvt4(v4f x) {
  v4bf r;
  r[0] = (__bf16)x[0]; r[1] = (__bf16)x[1];
  r[2] = (__bf16)x[2]; r[3] = (__bf16)x[3];
  return r;
}

// Async raw copy of 16 bytes global -> LDS (no VGPR round trip), ASYNCcnt-tracked.
__device__ __forceinline__ void async_load_lds_b128(void* lds_dst, const void* gsrc) {
  unsigned lds_off = (unsigned)(unsigned long long)(uintptr_t)lds_dst;  // low 32b = LDS addr
  asm volatile("global_load_async_to_lds_b128 %0, %1, off"
               :: "v"(lds_off), "v"(gsrc) : "memory");
}
__device__ __forceinline__ void wait_asynccnt0() {
  asm volatile("s_wait_asynccnt 0" ::: "memory");
}

// A-matrix fragment (16x32 bf16), LDS layout [m][ld], k offset kbase.
// ISA 7.12.2: lane L holds M=L%16; VGPR r: K = (r<4 ? 2r : 16+2(r-4)) + 8*(L/16).
__device__ __forceinline__ v16bf ldsA_frag(const __bf16* base, int row0, int ld, int kbase) {
  const int lane = threadIdx.x & 31;
  const int m = row0 + (lane & 15);
  const int h = lane >> 4;
  v16bf f;
#pragma unroll
  for (int r = 0; r < 8; ++r) {
    int k = kbase + ((r < 4) ? (2 * r) : (16 + 2 * (r - 4))) + 8 * h;
    f[2 * r]     = base[m * ld + k];
    f[2 * r + 1] = base[m * ld + k + 1];
  }
  return f;
}

// B-matrix fragment (32x16 bf16), LDS layout Bt[n][ld] (element (n,k)).
// ISA: lane L holds N=L%16; VGPR r: K = (r<4 ? 2r : 16+2(r-4)) + 8*(L/16).
__device__ __forceinline__ v16bf ldsB_fragT(const __bf16* base, int col0, int ld, int kbase) {
  const int lane = threadIdx.x & 31;
  const int n = col0 + (lane & 15);
  const int h = lane >> 4;
  v16bf f;
#pragma unroll
  for (int r = 0; r < 8; ++r) {
    int k = kbase + ((r < 4) ? (2 * r) : (16 + 2 * (r - 4))) + 8 * h;
    f[2 * r]     = base[n * ld + k];
    f[2 * r + 1] = base[n * ld + k + 1];
  }
  return f;
}

// B fragment from a row-major [k][ld] LDS tile (element (k,n)).
__device__ __forceinline__ v16bf ldsB_fragRM(const __bf16* base, int col0, int ld, int kbase) {
  const int lane = threadIdx.x & 31;
  const int n = col0 + (lane & 15);
  const int h = lane >> 4;
  v16bf f;
#pragma unroll
  for (int r = 0; r < 8; ++r) {
    int k = kbase + ((r < 4) ? (2 * r) : (16 + 2 * (r - 4))) + 8 * h;
    f[2 * r]     = base[k * ld + n];
    f[2 * r + 1] = base[(k + 1) * ld + n];
  }
  return f;
}

// A fragment built from fp32 LDS (softmax probabilities), bf16-converted inline.
__device__ __forceinline__ v16bf scoreA_frag(const float* sc, int row0, int ld, int kbase) {
  const int lane = threadIdx.x & 31;
  const int m = row0 + (lane & 15);
  const int h = lane >> 4;
  v16bf f;
#pragma unroll
  for (int r = 0; r < 8; ++r) {
    int k = kbase + ((r < 4) ? (2 * r) : (16 + 2 * (r - 4))) + 8 * h;
    f[2 * r]     = (__bf16)sc[m * ld + k];
    f[2 * r + 1] = (__bf16)sc[m * ld + k + 1];
  }
  return f;
}

// ---------------------------------------------------------------------------
// RMSNorm: one block per token, H=2048.
// ---------------------------------------------------------------------------
__global__ __launch_bounds__(256) void rmsnorm_kernel(const float* __restrict__ x,
                                                      const float* __restrict__ w,
                                                      float* __restrict__ out) {
  __shared__ float red[256];
  const int t = blockIdx.x;
  const float* xr = x + (size_t)t * 2048;
  float s = 0.f;
  for (int i = threadIdx.x; i < 2048; i += 256) { float v = xr[i]; s += v * v; }
  red[threadIdx.x] = s;
  __syncthreads();
  for (int off = 128; off > 0; off >>= 1) {
    if (threadIdx.x < off) red[threadIdx.x] += red[threadIdx.x + off];
    __syncthreads();
  }
  const float scale = rsqrtf(red[0] / 2048.f + 1e-5f);
  for (int i = threadIdx.x; i < 2048; i += 256)
    out[(size_t)t * 2048 + i] = w[i] * (xr[i] * scale);
}

// ---------------------------------------------------------------------------
// Generic bf16-WMMA GEMM: C[M,N] = A[M,K] @ B[K,N] (+ addend), fp32 in HBM.
// 128x128 tile per 256-thread block (8 waves), K-step 32.
// Staging: batched float4 loads (all in flight), then convert + LDS stores.
// ---------------------------------------------------------------------------
__global__ __launch_bounds__(256) void gemm_bf16_kernel(const float* __restrict__ A,
                                                        const float* __restrict__ B,
                                                        float* __restrict__ C,
                                                        const float* __restrict__ addend,
                                                        int M, int N, int K) {
  __shared__ __bf16 As[128 * LDA];  // [m][k]
  __shared__ __bf16 Bs[128 * LDA];  // Bt: [n][k]
  const int tid = threadIdx.x;
  const int bm = blockIdx.y * 128, bn = blockIdx.x * 128;
  const int wave = tid >> 5, lane = tid & 31;
  const int wm = (wave >> 1) * 32;
  const int wn = (wave & 1) * 64;
  v8f acc[2][4] = {};
  for (int k0 = 0; k0 < K; k0 += 32) {
    v4f ta[4], tb[4];
#pragma unroll
    for (int i = 0; i < 4; ++i) {   // issue all 8 b128 loads before any use
      int idx = i * 256 + tid;
      int r = idx >> 3, c4 = (idx & 7) * 4;
      ta[i] = *(const v4f*)&A[(size_t)(bm + r) * K + k0 + c4];
      int kk = idx >> 5, n4 = (idx & 31) * 4;
      tb[i] = *(const v4f*)&B[(size_t)(k0 + kk) * N + bn + n4];
    }
    if (k0 + 32 < K) {              // prefetch next weight K-slice
      int pr = tid >> 3, pc = (tid & 7) * 16;
      __builtin_prefetch(&B[(size_t)(k0 + 32 + pr) * N + bn + pc], 0, 0);
    }
#pragma unroll
    for (int i = 0; i < 4; ++i) {   // convert + store
      int idx = i * 256 + tid;
      int r = idx >> 3, c4 = (idx & 7) * 4;
      *(v4bf*)&As[r * LDA + c4] = cvt4(ta[i]);
      int kk = idx >> 5, n4 = (idx & 31) * 4;
      v4bf bb = cvt4(tb[i]);
#pragma unroll
      for (int j = 0; j < 4; ++j) Bs[(n4 + j) * LDA + kk] = bb[j];
    }
    __syncthreads();
    v16bf a0 = ldsA_frag(As, wm, LDA, 0);
    v16bf a1 = ldsA_frag(As, wm + 16, LDA, 0);
#pragma unroll
    for (int ni = 0; ni < 4; ++ni) {
      v16bf b = ldsB_fragT(Bs, wn + ni * 16, LDA, 0);
      acc[0][ni] = wmma_bf16(a0, b, acc[0][ni]);
      acc[1][ni] = wmma_bf16(a1, b, acc[1][ni]);
    }
    __syncthreads();
  }
  const int cn = lane & 15, hf = lane >> 4;
  if (addend != nullptr) {
#pragma unroll
    for (int mi = 0; mi < 2; ++mi)
#pragma unroll
      for (int ni = 0; ni < 4; ++ni)
#pragma unroll
        for (int r = 0; r < 8; ++r) {
          int row = bm + wm + mi * 16 + r + 8 * hf;
          int col = bn + wn + ni * 16 + cn;
          size_t o = (size_t)row * N + col;
          C[o] = acc[mi][ni][r] + addend[o];
        }
  } else {
#pragma unroll
    for (int mi = 0; mi < 2; ++mi)
#pragma unroll
      for (int ni = 0; ni < 4; ++ni)
#pragma unroll
        for (int r = 0; r < 8; ++r) {
          int row = bm + wm + mi * 16 + r + 8 * hf;
          int col = bn + wn + ni * 16 + cn;
          C[(size_t)row * N + col] = acc[mi][ni][r];
        }
  }
}

// ---------------------------------------------------------------------------
// RoPE in-place on q[S][16*128] and k[S][4*128] (cache_index == 0).
// ---------------------------------------------------------------------------
__global__ __launch_bounds__(256) void rope_kernel(float* __restrict__ q,
                                                   float* __restrict__ k,
                                                   const float* __restrict__ cosb,
                                                   const float* __restrict__ sinb) {
  const int t = blockIdx.x;
  const float* cr = cosb + (size_t)t * 128;
  const float* sr = sinb + (size_t)t * 128;
  for (int p = threadIdx.x; p < 16 * 64; p += 256) {
    int hh = p >> 6, d = p & 63;
    float* base = q + (size_t)t * 2048 + hh * 128;
    float a = base[d], b = base[d + 64];
    base[d]      = a * cr[d]      - b * sr[d];
    base[d + 64] = b * cr[d + 64] + a * sr[d + 64];
  }
  for (int p = threadIdx.x; p < 4 * 64; p += 256) {
    int hh = p >> 6, d = p & 63;
    float* base = k + (size_t)t * 512 + hh * 128;
    float a = base[d], b = base[d + 64];
    base[d]      = a * cr[d]      - b * sr[d];
    base[d + 64] = b * cr[d + 64] + a * sr[d + 64];
  }
}

// ---------------------------------------------------------------------------
// Attention: one block per (head, 64-query tile). Score rows (64 x 1028 fp32)
// + Q tile + K/V staging in 291KB dynamic LDS (< 320KB/WGP). GQA: head h uses
// KV head h/4. Causal mask, full-row softmax, P@V via WMMA.
// ---------------------------------------------------------------------------
__global__ __launch_bounds__(256) void attn_kernel(const float* __restrict__ q,
                                                   const float* __restrict__ k,
                                                   const float* __restrict__ v,
                                                   float* __restrict__ o) {
  extern __shared__ char smem[];
  float*  sc  = (float*)smem;                              // 64 x LDS_SC fp32
  __bf16* Qs  = (__bf16*)(smem + 64 * LDS_SC * 4);         // 64 x LDQ
  __bf16* KVs = Qs + 64 * LDQ;                             // 64 x LDQ
  const int head = blockIdx.x, qt = blockIdx.y;
  const int kvh = head >> 2;
  const int tid = threadIdx.x, wave = tid >> 5, lane = tid & 31;
  const float scale = 0.08838834764831845f;                // 1/sqrt(128)

  // Q tile: batched float4 loads + bf16x4 stores.
  {
    v4f tq[8];
#pragma unroll
    for (int i = 0; i < 8; ++i) {
      int idx = i * 256 + tid, r = idx >> 5, d4 = (idx & 31) * 4;
      tq[i] = *(const v4f*)&q[(size_t)(qt * 64 + r) * 2048 + head * 128 + d4];
    }
#pragma unroll
    for (int i = 0; i < 8; ++i) {
      int idx = i * 256 + tid, r = idx >> 5, d4 = (idx & 31) * 4;
      *(v4bf*)&Qs[r * LDQ + d4] = cvt4(tq[i]);
    }
  }
  __syncthreads();

  // ---- Phase 1: scores = Q K^T * scale, 16 key blocks of 64 ----
  const int wm1 = (wave & 3) * 16, wn1 = (wave >> 2) * 32;
  for (int kc = 0; kc < 16; ++kc) {
    {
      v4f tk[8];
#pragma unroll
      for (int i = 0; i < 8; ++i) {
        int idx = i * 256 + tid, r = idx >> 5, d4 = (idx & 31) * 4;
        tk[i] = *(const v4f*)&k[(size_t)(kc * 64 + r) * 512 + kvh * 128 + d4];
      }
#pragma unroll
      for (int i = 0; i < 8; ++i) {
        int idx = i * 256 + tid, r = idx >> 5, d4 = (idx & 31) * 4;
        *(v4bf*)&KVs[r * LDQ + d4] = cvt4(tk[i]);
      }
    }
    __syncthreads();
    v8f a2[2] = {};
#pragma unroll
    for (int kk = 0; kk < 4; ++kk) {                       // contraction over HD=128
      v16bf a = ldsA_frag(Qs, wm1, LDQ, kk * 32);
#pragma unroll
      for (int ni = 0; ni < 2; ++ni) {
        v16bf b = ldsB_fragT(KVs, wn1 + ni * 16, LDQ, kk * 32);
        a2[ni] = wmma_bf16(a, b, a2[ni]);
      }
    }
    {
      const int cn = lane & 15, hf = lane >> 4;
#pragma unroll
      for (int ni = 0; ni < 2; ++ni)
#pragma unroll
        for (int r = 0; r < 8; ++r)
          sc[(wm1 + r + 8 * hf) * LDS_SC + kc * 64 + wn1 + ni * 16 + cn] = a2[ni][r] * scale;
    }
    __syncthreads();
  }

  // ---- Phase 2: causal softmax, 8 rows per wave, wave32 shuffles ----
  for (int rr = 0; rr < 8; ++rr) {
    const int row = wave * 8 + rr;
    const int qglob = qt * 64 + row;
    float mx = -3.0e38f;
    for (int j = lane; j < 1024; j += 32) {
      float vv = (j <= qglob) ? sc[row * LDS_SC + j] : -3.0e38f;
      mx = fmaxf(mx, vv);
    }
#pragma unroll
    for (int off = 16; off > 0; off >>= 1) mx = fmaxf(mx, __shfl_xor(mx, off, 32));
    float sum = 0.f;
    for (int j = lane; j < 1024; j += 32) {
      float e = (j <= qglob) ? __expf(sc[row * LDS_SC + j] - mx) : 0.f;
      sc[row * LDS_SC + j] = e;
      sum += e;
    }
#pragma unroll
    for (int off = 16; off > 0; off >>= 1) sum += __shfl_xor(sum, off, 32);
    const float inv = 1.f / sum;
    for (int j = lane; j < 1024; j += 32) sc[row * LDS_SC + j] *= inv;
  }
  __syncthreads();

  // ---- Phase 3: O = P @ V ----
  const int wm3 = (wave & 3) * 16, wn3 = (wave >> 2) * 64;
  v8f accO[4] = {};
  for (int kc = 0; kc < 16; ++kc) {
    {
      v4f tv[8];
#pragma unroll
      for (int i = 0; i < 8; ++i) {
        int idx = i * 256 + tid, r = idx >> 5, d4 = (idx & 31) * 4;
        tv[i] = *(const v4f*)&v[(size_t)(kc * 64 + r) * 512 + kvh * 128 + d4];
      }
#pragma unroll
      for (int i = 0; i < 8; ++i) {
        int idx = i * 256 + tid, r = idx >> 5, d4 = (idx & 31) * 4;
        *(v4bf*)&KVs[r * LDQ + d4] = cvt4(tv[i]);
      }
    }
    __syncthreads();
#pragma unroll
    for (int kk = 0; kk < 2; ++kk) {                       // 64 keys = 2 K-steps
      v16bf a = scoreA_frag(sc, wm3, LDS_SC, kc * 64 + kk * 32);
#pragma unroll
      for (int ni = 0; ni < 4; ++ni) {
        v16bf b = ldsB_fragRM(KVs, wn3 + ni * 16, LDQ, kk * 32);
        accO[ni] = wmma_bf16(a, b, accO[ni]);
      }
    }
    __syncthreads();
  }
  const int cn = lane & 15, hf = lane >> 4;
#pragma unroll
  for (int ni = 0; ni < 4; ++ni)
#pragma unroll
    for (int r = 0; r < 8; ++r) {
      int row = qt * 64 + wm3 + r + 8 * hf;
      int col = wn3 + ni * 16 + cn;
      o[(size_t)row * 2048 + head * 128 + col] = accO[ni][r];
    }
}

// ---------------------------------------------------------------------------
// MoE gating: logits = x2 @ gate_w (2048x8), softmax, top-2, renormalize.
// ---------------------------------------------------------------------------
__global__ __launch_bounds__(256) void gating_kernel(const float* __restrict__ x2,
                                                     const float* __restrict__ gw,
                                                     float* __restrict__ dw) {
  __shared__ float red[256];
  __shared__ float logits[8];
  const int t = blockIdx.x;
  float acc8[8] = {};
  for (int i = threadIdx.x; i < 2048; i += 256) {
    float xv = x2[(size_t)t * 2048 + i];
#pragma unroll
    for (int e = 0; e < 8; ++e) acc8[e] += xv * gw[i * 8 + e];
  }
  for (int e = 0; e < 8; ++e) {
    red[threadIdx.x] = acc8[e];
    __syncthreads();
    for (int off = 128; off > 0; off >>= 1) {
      if (threadIdx.x < off) red[threadIdx.x] += red[threadIdx.x + off];
      __syncthreads();
    }
    if (threadIdx.x == 0) logits[e] = red[0];
    __syncthreads();
  }
  if (threadIdx.x == 0) {
    float mx = logits[0];
    for (int e = 1; e < 8; ++e) mx = fmaxf(mx, logits[e]);
    float pe[8], s = 0.f;
    for (int e = 0; e < 8; ++e) { pe[e] = __expf(logits[e] - mx); s += pe[e]; }
    for (int e = 0; e < 8; ++e) pe[e] /= s;
    int i1 = 0;
    for (int e = 1; e < 8; ++e) if (pe[e] > pe[i1]) i1 = e;
    int i2 = (i1 == 0) ? 1 : 0;
    for (int e = 0; e < 8; ++e) if (e != i1 && pe[e] > pe[i2]) i2 = e;
    const float wsum = pe[i1] + pe[i2];
    for (int e = 0; e < 8; ++e)
      dw[(size_t)t * 8 + e] = (e == i1) ? pe[i1] / wsum : ((e == i2) ? pe[i2] / wsum : 0.f);
  }
}

// ---------------------------------------------------------------------------
// MoE up: h[e,t,f] = silu(x2 @ up_w[e]) * (x2 @ gatep_w[e]), stored bf16.
// 64 tokens x 128 F per block; waves 0-3 compute `up`, 4-7 `gatep`,
// results cross via LDS. Weight streams prefetched one K-slice ahead.
// ---------------------------------------------------------------------------
__global__ __launch_bounds__(256) void moe_up_kernel(const float* __restrict__ x2,
                                                     const float* __restrict__ upw,
                                                     const float* __restrict__ gpw,
                                                     __bf16* __restrict__ h) {
  __shared__ __bf16 As[64 * LDA];
  __shared__ __bf16 Bus[128 * LDA];
  __shared__ __bf16 Bgs[128 * LDA];
  __shared__ float  hbuf[64 * 128];
  const int bn = blockIdx.x * 128;
  const int bm = blockIdx.y * 64;
  const int e  = blockIdx.z;
  const int tid = threadIdx.x, wave = tid >> 5, lane = tid & 31;
  const int grp = wave >> 2;         // 0 = up, 1 = gatep
  const int wm  = (wave & 3) * 16;
  const size_t wbase = (size_t)e * 2048 * 7168;
  v8f acc[8] = {};
  for (int k0 = 0; k0 < 2048; k0 += 32) {
    v4f ta[2], tu[4], tg[4];
#pragma unroll
    for (int i = 0; i < 2; ++i) {
      int idx = i * 256 + tid, r = idx >> 3, c4 = (idx & 7) * 4;
      ta[i] = *(const v4f*)&x2[(size_t)(bm + r) * 2048 + k0 + c4];
    }
#pragma unroll
    for (int i = 0; i < 4; ++i) {
      int idx = i * 256 + tid, kk = idx >> 5, n4 = (idx & 31) * 4;
      tu[i] = *(const v4f*)&upw[wbase + (size_t)(k0 + kk) * 7168 + bn + n4];
      tg[i] = *(const v4f*)&gpw[wbase + (size_t)(k0 + kk) * 7168 + bn + n4];
    }
    if (k0 + 32 < 2048) {
      int pr = tid >> 3, pc = (tid & 7) * 16;
      __builtin_prefetch(&upw[wbase + (size_t)(k0 + 32 + pr) * 7168 + bn + pc], 0, 0);
      __builtin_prefetch(&gpw[wbase + (size_t)(k0 + 32 + pr) * 7168 + bn + pc], 0, 0);
    }
#pragma unroll
    for (int i = 0; i < 2; ++i) {
      int idx = i * 256 + tid, r = idx >> 3, c4 = (idx & 7) * 4;
      *(v4bf*)&As[r * LDA + c4] = cvt4(ta[i]);
    }
#pragma unroll
    for (int i = 0; i < 4; ++i) {
      int idx = i * 256 + tid, kk = idx >> 5, n4 = (idx & 31) * 4;
      v4bf ub = cvt4(tu[i]), gb = cvt4(tg[i]);
#pragma unroll
      for (int j = 0; j < 4; ++j) {
        Bus[(n4 + j) * LDA + kk] = ub[j];
        Bgs[(n4 + j) * LDA + kk] = gb[j];
      }
    }
    __syncthreads();
    const __bf16* Bsel = grp ? Bgs : Bus;
    v16bf a = ldsA_frag(As, wm, LDA, 0);
#pragma unroll
    for (int ni = 0; ni < 8; ++ni) {
      v16bf b = ldsB_fragT(Bsel, ni * 16, LDA, 0);
      acc[ni] = wmma_bf16(a, b, acc[ni]);
    }
    __syncthreads();
  }
  const int cn = lane & 15, hf = lane >> 4;
  if (grp == 1) {
#pragma unroll
    for (int ni = 0; ni < 8; ++ni)
#pragma unroll
      for (int r = 0; r < 8; ++r)
        hbuf[(wm + r + 8 * hf) * 128 + ni * 16 + cn] = acc[ni][r];
  }
  __syncthreads();
  if (grp == 0) {
#pragma unroll
    for (int ni = 0; ni < 8; ++ni)
#pragma unroll
      for (int r = 0; r < 8; ++r) {
        int row = wm + r + 8 * hf, col = ni * 16 + cn;
        float u  = acc[ni][r];
        float g  = hbuf[row * 128 + col];
        float hv = (u / (1.f + __expf(-u))) * g;     // silu(u) * g
        h[((size_t)e * 1024 + bm + row) * 7168 + bn + col] = (__bf16)hv;
      }
  }
}

// ---------------------------------------------------------------------------
// MoE down + combine: out = resid + sum_e dense_w[:,e] * (h[e] @ down_w[e]).
// Experts with all-zero routing weight in the token tile are skipped (~2x).
// A tiles (already bf16 in HBM) are copied with global_load_async_to_lds_b128.
// ---------------------------------------------------------------------------
__global__ __launch_bounds__(256) void moe_down_kernel(const __bf16* __restrict__ h,
                                                       const float* __restrict__ dww,
                                                       const float* __restrict__ dw,
                                                       const float* __restrict__ resid,
                                                       float* __restrict__ out) {
  __shared__ __bf16 As[128 * LDA];
  __shared__ __bf16 Bs[128 * LDA];
  __shared__ float  wts[128 * 8];
  __shared__ int    anyflag;
  const int bn = blockIdx.x * 128;
  const int bm = blockIdx.y * 128;
  const int tid = threadIdx.x, wave = tid >> 5, lane = tid & 31;
  const int wm = (wave >> 1) * 32, wn = (wave & 1) * 64;
  for (int i = tid; i < 128 * 8; i += 256)
    wts[i] = dw[(size_t)(bm + (i >> 3)) * 8 + (i & 7)];
  v8f tot[2][4] = {};
  for (int e = 0; e < 8; ++e) {
    __syncthreads();
    if (tid == 0) anyflag = 0;
    __syncthreads();
    if (tid < 128 && wts[tid * 8 + e] != 0.f) anyflag = 1;
    __syncthreads();
    if (!anyflag) continue;                       // no token in tile routed to e
    v8f acc[2][4] = {};
    for (int k0 = 0; k0 < 7168; k0 += 32) {
      // A: 128x32 bf16 tile, raw async copy (16B chunks), no VGPR staging.
#pragma unroll
      for (int i = 0; i < 2; ++i) {
        int idx = i * 256 + tid;
        int r = idx >> 2, c8 = (idx & 3) * 8;     // 4x 16B chunks per 64B row
        async_load_lds_b128(&As[r * LDA + c8],
                            &h[((size_t)e * 1024 + bm + r) * 7168 + k0 + c8]);
      }
      // B: fp32 -> bf16 conversion staging, batched float4 loads.
      v4f tb[4];
#pragma unroll
      for (int i = 0; i < 4; ++i) {
        int idx = i * 256 + tid, kk = idx >> 5, n4 = (idx & 31) * 4;
        tb[i] = *(const v4f*)&dww[(size_t)e * 7168 * 2048 + (size_t)(k0 + kk) * 2048 + bn + n4];
      }
      if (k0 + 32 < 7168) {
        int pr = tid >> 3, pc = (tid & 7) * 16;
        __builtin_prefetch(&dww[(size_t)e * 7168 * 2048 + (size_t)(k0 + 32 + pr) * 2048 + bn + pc], 0, 0);
      }
#pragma unroll
      for (int i = 0; i < 4; ++i) {
        int idx = i * 256 + tid, kk = idx >> 5, n4 = (idx & 31) * 4;
        v4bf bb = cvt4(tb[i]);
#pragma unroll
        for (int j = 0; j < 4; ++j) Bs[(n4 + j) * LDA + kk] = bb[j];
      }
      wait_asynccnt0();
      __syncthreads();
      v16bf a0 = ldsA_frag(As, wm, LDA, 0);
      v16bf a1 = ldsA_frag(As, wm + 16, LDA, 0);
#pragma unroll
      for (int ni = 0; ni < 4; ++ni) {
        v16bf b = ldsB_fragT(Bs, wn + ni * 16, LDA, 0);
        acc[0][ni] = wmma_bf16(a0, b, acc[0][ni]);
        acc[1][ni] = wmma_bf16(a1, b, acc[1][ni]);
      }
      __syncthreads();
    }
    const int hf = lane >> 4;
#pragma unroll
    for (int mi = 0; mi < 2; ++mi)
#pragma unroll
      for (int ni = 0; ni < 4; ++ni)
#pragma unroll
        for (int r = 0; r < 8; ++r) {
          int row = wm + mi * 16 + r + 8 * hf;
          tot[mi][ni][r] += wts[row * 8 + e] * acc[mi][ni][r];
        }
  }
  const int cn = lane & 15, hf = lane >> 4;
#pragma unroll
  for (int mi = 0; mi < 2; ++mi)
#pragma unroll
    for (int ni = 0; ni < 4; ++ni)
#pragma unroll
      for (int r = 0; r < 8; ++r) {
        int row = bm + wm + mi * 16 + r + 8 * hf;
        int col = bn + wn + ni * 16 + cn;
        size_t o = (size_t)row * 2048 + col;
        out[o] = resid[o] + tot[mi][ni][r];
      }
}

// ---------------------------------------------------------------------------
// Host launcher.
// ---------------------------------------------------------------------------
extern "C" void kernel_launch(void* const* d_in, const int* in_sizes, int n_in,
                              void* d_out, int out_size, void* d_ws, size_t ws_size,
                              hipStream_t stream) {
  (void)in_sizes; (void)n_in; (void)out_size; (void)ws_size;
  const float* hidden = (const float*)d_in[0];
  const float* cosb   = (const float*)d_in[1];
  const float* sinb   = (const float*)d_in[2];
  // d_in[3], d_in[4]: zero KV caches (cache_index == 0, cache == fresh k/v)
  const float* q_w    = (const float*)d_in[5];
  const float* k_w    = (const float*)d_in[6];
  const float* v_w    = (const float*)d_in[7];
  const float* o_w    = (const float*)d_in[8];
  const float* ln1    = (const float*)d_in[9];
  const float* ln2    = (const float*)d_in[10];
  const float* gate_w = (const float*)d_in[11];
  const float* up_w   = (const float*)d_in[12];
  const float* gp_w   = (const float*)d_in[13];
  const float* down_w = (const float*)d_in[14];
  // d_in[15]: attention_mask (all true), d_in[16]: cache_index (0)
  float* out = (float*)d_out;

  char* ws = (char*)d_ws;
  size_t off = 0;
  auto carve = [&](size_t bytes) -> char* {
    char* p = ws + off;
    off += (bytes + 255) & ~(size_t)255;
    return p;
  };
  float*  xn   = (float*)carve(1024ull * 2048 * 4);
  float*  qb   = (float*)carve(1024ull * 2048 * 4);
  float*  kb   = (float*)carve(1024ull * 512 * 4);
  float*  vb   = (float*)carve(1024ull * 512 * 4);
  float*  attn = (float*)carve(1024ull * 2048 * 4);
  float*  xr2  = (float*)carve(1024ull * 2048 * 4);
  float*  x2   = (float*)carve(1024ull * 2048 * 4);
  float*  dwts = (float*)carve(1024ull * 8 * 4);
  __bf16* hbuf = (__bf16*)carve(8ull * 1024 * 7168 * 2);

  rmsnorm_kernel<<<1024, 256, 0, stream>>>(hidden, ln1, xn);
  gemm_bf16_kernel<<<dim3(16, 8), 256, 0, stream>>>(xn, q_w, qb, nullptr, 1024, 2048, 2048);
  gemm_bf16_kernel<<<dim3(4, 8),  256, 0, stream>>>(xn, k_w, kb, nullptr, 1024, 512, 2048);
  gemm_bf16_kernel<<<dim3(4, 8),  256, 0, stream>>>(xn, v_w, vb, nullptr, 1024, 512, 2048);
  rope_kernel<<<1024, 256, 0, stream>>>(qb, kb, cosb, sinb);

  const size_t attn_lds = 64 * LDS_SC * 4 + 2 * 64 * LDQ * 2;  // 297984 B < 320KB/WGP
  hipFuncSetAttribute(reinterpret_cast<const void*>(attn_kernel),
                      hipFuncAttributeMaxDynamicSharedMemorySize, (int)attn_lds);
  attn_kernel<<<dim3(16, 16), 256, attn_lds, stream>>>(qb, kb, vb, attn);

  gemm_bf16_kernel<<<dim3(16, 8), 256, 0, stream>>>(attn, o_w, xr2, hidden, 1024, 2048, 2048);
  rmsnorm_kernel<<<1024, 256, 0, stream>>>(xr2, ln2, x2);
  gating_kernel<<<1024, 256, 0, stream>>>(x2, gate_w, dwts);
  moe_up_kernel<<<dim3(56, 16, 8), 256, 0, stream>>>(x2, up_w, gp_w, hbuf);
  moe_down_kernel<<<dim3(16, 8), 256, 0, stream>>>(hbuf, down_w, dwts, xr2, out);
}
